// SubjectConditionedLayer_89489938579611
// MI455X (gfx1250) — compile-verified, matrix operand order checked by reading, not run on hardware
//
#include <hip/hip_runtime.h>

#define IN_F   1024
#define OUT_F  1024
#define RANK   4
#define NUM_AD 10
#define BATCH  32
#define SEQ    2048
#define SCALE  0.25f            // ALPHA / RANK

#define TILE_M 128
#define TILE_N 128
#define TILE_K 32
#define NKT    (IN_F / TILE_K)  // 32 k-steps
#define LDSS   40               // LDS row stride in bf16 elems (80B: conflict-free b128 frags)

typedef __attribute__((ext_vector_type(16))) __bf16        v16bf;
typedef __attribute__((ext_vector_type(2)))  __bf16        v2bf;
typedef __attribute__((ext_vector_type(8)))  float         v8f;
typedef __attribute__((ext_vector_type(4)))  unsigned int  u32x4;
typedef __attribute__((ext_vector_type(2)))  unsigned int  u32x2;
typedef __attribute__((ext_vector_type(4)))  float         f32x4;

union FragU { v16bf v; u32x4 q[2]; };

static __device__ __forceinline__ unsigned int pk_bf16(float lo, float hi) {
#if __has_builtin(__builtin_amdgcn_cvt_pk_bf16_f32)
  union { v2bf v; unsigned int u; } cv;
  cv.v = __builtin_amdgcn_cvt_pk_bf16_f32(lo, hi);
  return cv.u;
#elif __has_builtin(__builtin_amdgcn_perm)
  // round-to-nearest (ties away) then pack both hi16 halves in ONE v_perm_b32:
  // result bytes = { lo[2], lo[3], hi[2], hi[3] }  -> selector 0x07060302
  const unsigned int ul = __float_as_uint(lo) + 0x8000u;
  const unsigned int uh = __float_as_uint(hi) + 0x8000u;
  return __builtin_amdgcn_perm(uh, ul, 0x07060302u);
#else
  unsigned int ul = __float_as_uint(lo);
  unsigned int uh = __float_as_uint(hi);
  ul = (ul + 0x7FFFu + ((ul >> 16) & 1u)) >> 16;
  uh = (uh + 0x7FFFu + ((uh >> 16) & 1u)) >> 16;
  return (ul & 0xFFFFu) | (uh << 16);
#endif
}

// CDNA5 async DMA: global -> LDS, 32 bytes per lane (two b128), no VGPR staging.
// IOFFSET is added to BOTH the LDS and global addresses per the ISA.
#define ASYNC_LDS_LOAD_32B(ldsoff_u32, gaddr_u64)                              \
  asm volatile("global_load_async_to_lds_b128 %0, %1, off\n\t"                 \
               "global_load_async_to_lds_b128 %0, %1, off offset:16"           \
               :: "v"(ldsoff_u32), "v"(gaddr_u64) : "memory")

#define WAIT_ASYNCCNT_0() asm volatile("s_wait_asynccnt 0x0" ::: "memory")

// ---------------------------------------------------------------------------
// Kernel 1: fold LoRA into per-adapter effective weights (bf16):
//   W_eff[a,o,i] = W[o,i] + 0.25 * sum_r B[a,o,r] * A[a,r,i]
// ---------------------------------------------------------------------------
__global__ __launch_bounds__(256) void build_weff_kernel(
    const float* __restrict__ W, const float* __restrict__ lA,
    const float* __restrict__ lB, unsigned short* __restrict__ weff) {
  const int a   = blockIdx.y;
  const int t   = blockIdx.x * 256 + threadIdx.x;
  const int idx = t * 4;                 // element index in [OUT_F*IN_F)
  const int o   = idx >> 10;             // / IN_F
  const int i   = idx & (IN_F - 1);

  f32x4 w = *(const f32x4*)(W + idx);
  float a0 = w.x, a1 = w.y, a2 = w.z, a3 = w.w;
#pragma unroll
  for (int r = 0; r < RANK; ++r) {
    const float brs = lB[(a * OUT_F + o) * RANK + r] * SCALE;
    const f32x4 av  = *(const f32x4*)(lA + ((a * RANK + r) * IN_F) + i);
    a0 += brs * av.x; a1 += brs * av.y; a2 += brs * av.z; a3 += brs * av.w;
  }
  u32x2 pk;
  pk.x = pk_bf16(a0, a1);
  pk.y = pk_bf16(a2, a3);
  *(u32x2*)(weff + (size_t)a * OUT_F * IN_F + idx) = pk;
}

// ---------------------------------------------------------------------------
// Kernel 2: batched GEMM  out[b,s,:] = x[b,s,:] @ W_eff[sid[b]]^T + bias
// 128x128 tile / 256 threads (8 wave32, 4x2 waves, each wave 2x4 WMMA tiles)
// Double-buffered LDS, async-to-LDS W tiles, one barrier per k-step.
// ---------------------------------------------------------------------------
__global__ __launch_bounds__(256) void lora_gemm_kernel(
    const float* __restrict__ x, const unsigned short* __restrict__ weff,
    const float* __restrict__ bias, const int* __restrict__ sid,
    float* __restrict__ out) {
  __shared__ unsigned short lx[2][TILE_M * LDSS];   // 2 x 10 KB
  __shared__ unsigned short lw[2][TILE_N * LDSS];   // 2 x 10 KB

  const int nt   = blockIdx.x;            // 0..OUT_F/128-1
  const int mt   = blockIdx.y;            // 0..B*S/128-1
  const int tid  = threadIdx.x;
  const int lane = tid & 31;
  const int wave = tid >> 5;              // 0..7
  const int wm   = wave >> 1;             // 0..3  (M strip of 32 rows)
  const int wn   = wave & 1;              // 0..1  (N strip of 64 cols)

  const int b = mt >> 4;                  // (mt*128)/SEQ ; SEQ/TILE_M == 16
  const unsigned short* Wb = weff + (size_t)sid[b] * (OUT_F * IN_F);

  // cooperative tile loads: thread -> (row, 16-elem k-segment)
  const int lr = tid >> 1;                // 0..127
  const int lk = (tid & 1) << 4;          // 0 or 16

  const float*          xg = x  + (size_t)(mt * TILE_M + lr) * IN_F + lk;
  const unsigned short* wg = Wb + (size_t)(nt * TILE_N + lr) * IN_F + lk;

  // per-thread staging addresses for both LDS buffers
  unsigned short* lxs[2] = { &lx[0][lr * LDSS + lk], &lx[1][lr * LDSS + lk] };
  const unsigned int lwd[2] = {
    (unsigned int)(uintptr_t)&lw[0][lr * LDSS + lk],
    (unsigned int)(uintptr_t)&lw[1][lr * LDSS + lk] };

  // per-lane fragment base pointers (ISA 16-bit A-matrix layout)
  const int hf = lane >> 4;               // K-half select
  const int rr = lane & 15;               // row/col within 16
  const unsigned short* axb[2] = {
    &lx[0][(wm * 32 + rr) * LDSS + hf * 8], &lx[1][(wm * 32 + rr) * LDSS + hf * 8] };
  const unsigned short* bwb[2] = {
    &lw[0][(wn * 64 + rr) * LDSS + hf * 8], &lw[1][(wn * 64 + rr) * LDSS + hf * 8] };

  v8f acc[2][4];
#pragma unroll
  for (int i = 0; i < 2; ++i)
#pragma unroll
    for (int j = 0; j < 4; ++j) acc[i][j] = (v8f){};

  // ---- prologue: stage k-tile 0 into buffer 0 ----
  ASYNC_LDS_LOAD_32B(lwd[0], (unsigned long long)(uintptr_t)wg);
  {
    f32x4 xa0 = *(const f32x4*)(xg + 0);
    f32x4 xa1 = *(const f32x4*)(xg + 4);
    f32x4 xa2 = *(const f32x4*)(xg + 8);
    f32x4 xa3 = *(const f32x4*)(xg + 12);
    u32x4 xp;
    xp.x = pk_bf16(xa0.x, xa0.y); xp.y = pk_bf16(xa0.z, xa0.w);
    xp.z = pk_bf16(xa1.x, xa1.y); xp.w = pk_bf16(xa1.z, xa1.w);
    *(u32x4*)(lxs[0] + 0) = xp;
    xp.x = pk_bf16(xa2.x, xa2.y); xp.y = pk_bf16(xa2.z, xa2.w);
    xp.z = pk_bf16(xa3.x, xa3.y); xp.w = pk_bf16(xa3.z, xa3.w);
    *(u32x4*)(lxs[0] + 8) = xp;
  }
  WAIT_ASYNCCNT_0();
  __syncthreads();

  // ---- main loop: one barrier per k-step ----
#pragma unroll 2
  for (int kt = 0; kt < NKT; ++kt) {
    const int cur = kt & 1;
    const int nxt = cur ^ 1;

    f32x4 xa0, xa1, xa2, xa3;
    if (kt + 1 < NKT) {
      // issue next W tile DMA (global -> LDS, alternate buffer)
      ASYNC_LDS_LOAD_32B(lwd[nxt],
          (unsigned long long)(uintptr_t)(wg + (kt + 1) * TILE_K));
      // issue next x tile global loads into registers
      const float* xn2 = xg + (kt + 1) * TILE_K;
      xa0 = *(const f32x4*)(xn2 + 0);
      xa1 = *(const f32x4*)(xn2 + 4);
      xa2 = *(const f32x4*)(xn2 + 8);
      xa3 = *(const f32x4*)(xn2 + 12);
    }

    // compute from current buffers
    v16bf af[2], bfr[4];
#pragma unroll
    for (int i = 0; i < 2; ++i) {
      const unsigned short* p = axb[cur] + (i * 16) * LDSS;
      FragU fu;
      fu.q[0] = *(const u32x4*)(p);
      fu.q[1] = *(const u32x4*)(p + 16);
      af[i] = fu.v;
    }
#pragma unroll
    for (int j = 0; j < 4; ++j) {
      const unsigned short* p = bwb[cur] + (j * 16) * LDSS;
      FragU fu;
      fu.q[0] = *(const u32x4*)(p);
      fu.q[1] = *(const u32x4*)(p + 16);
      bfr[j] = fu.v;
    }
#pragma unroll
    for (int i = 0; i < 2; ++i)
#pragma unroll
      for (int j = 0; j < 4; ++j)
        acc[i][j] = __builtin_amdgcn_wmma_f32_16x16x32_bf16(
            false, af[i], false, bfr[j], (short)0, acc[i][j], false, false);

    if (kt + 1 < NKT) {
      // pack next x tile (1x v_perm_b32 per bf16 pair) into alternate buffer
      u32x4 xp;
      xp.x = pk_bf16(xa0.x, xa0.y); xp.y = pk_bf16(xa0.z, xa0.w);
      xp.z = pk_bf16(xa1.x, xa1.y); xp.w = pk_bf16(xa1.z, xa1.w);
      *(u32x4*)(lxs[nxt] + 0) = xp;
      xp.x = pk_bf16(xa2.x, xa2.y); xp.y = pk_bf16(xa2.z, xa2.w);
      xp.z = pk_bf16(xa3.x, xa3.y); xp.w = pk_bf16(xa3.z, xa3.w);
      *(u32x4*)(lxs[nxt] + 8) = xp;
    }

    WAIT_ASYNCCNT_0();   // next-buffer W DMA complete (latency hidden by wmma)
    __syncthreads();     // next-buffer x stores visible; cur buffer free
  }

  // epilogue: + bias, store per 16x16 f32 C/D layout (VGPR v -> row v + 8*hf)
  const int col0 = nt * TILE_N + wn * 64;
  const int row0 = mt * TILE_M + wm * 32;
#pragma unroll
  for (int j = 0; j < 4; ++j) {
    const int col = col0 + j * 16 + rr;
    const float bv = bias[col];
#pragma unroll
    for (int i = 0; i < 2; ++i) {
#pragma unroll
      for (int v = 0; v < 8; ++v) {
        const int gr = row0 + i * 16 + hf * 8 + v;
        out[(size_t)gr * OUT_F + col] = acc[i][j][v] + bv;
      }
    }
  }
}

// ---------------------------------------------------------------------------
extern "C" void kernel_launch(void* const* d_in, const int* in_sizes, int n_in,
                              void* d_out, int out_size, void* d_ws, size_t ws_size,
                              hipStream_t stream) {
  const float* x   = (const float*)d_in[0];
  const float* W   = (const float*)d_in[1];
  const float* bs  = (const float*)d_in[2];
  const float* lA  = (const float*)d_in[3];
  const float* lB  = (const float*)d_in[4];
  const int*   sid = (const int*)d_in[5];
  float* out = (float*)d_out;

  // workspace: 10 adapters x 1024x1024 bf16 = 20 MB of effective weights
  unsigned short* weff = (unsigned short*)d_ws;

  build_weff_kernel<<<dim3((OUT_F * IN_F) / (4 * 256), NUM_AD), 256, 0, stream>>>(
      W, lA, lB, weff);
  lora_gemm_kernel<<<dim3(OUT_F / TILE_N, (BATCH * SEQ) / TILE_M), 256, 0, stream>>>(
      x, weff, bs, sid, out);
}
